// GlobalAttention_16527034155053
// MI455X (gfx1250) — matrix operand order repaired and checked by reading, not compiled, not run on hardware
//
#include <hip/hip_runtime.h>
#include <hip/hip_bf16.h>

// ---------------------------------------------------------------------------
// GlobalAttention (dilated-window MHA) for MI455X / gfx1250, wave32 WMMA.
// GEMMs: 256x128 block tile, 64x64 wave tile, K staged 64-wide in LDS,
// v_wmma_f32_16x16x32_f16 with f32 accumulation throughout.
// ---------------------------------------------------------------------------

typedef _Float16 v16h __attribute__((ext_vector_type(16)));
typedef float    v8f  __attribute__((ext_vector_type(8)));
typedef _Float16 h8   __attribute__((ext_vector_type(8)));
typedef _Float16 h4   __attribute__((ext_vector_type(4)));

#define D_MODEL  1024
#define N_HEADS  16
#define HEAD_DIM 64
#define SEQ_T    8192
#define BATCH    4
#define WDIL     64
#define NWIN     128
#define NTOK     (BATCH * SEQ_T)   // 32768 tokens

// ln(10000)/32
#define ROPE_LOGK 0.28782313662425575f

// ---- WMMA fragment loaders (wave32, 16x16x32 f16) --------------------------
// A-matrix: lane L -> row (L&15); halfs e<8: K = (L>>4)*8 + e ; e>=8: K = 16 + (L>>4)*8 + (e-8)
__device__ __forceinline__ v16h frag_a(const _Float16* base, int row, int ld, int kb) {
    const int lane = threadIdx.x & 31;
    const _Float16* p = base + (size_t)(row + (lane & 15)) * ld + kb + ((lane >> 4) << 3);
    v16h r;
    ((h8*)&r)[0] = *(const h8*)(p);
    ((h8*)&r)[1] = *(const h8*)(p + 16);
    return r;
}
// B-matrix (stored as [n][k]): lane L -> col (L&15); halfs e<8: K = (L>>4)*8 + e ; e>=8: K = 16 + (L>>4)*8 + (e-8)
__device__ __forceinline__ v16h frag_b(const _Float16* base, int col, int ld, int kb) {
    const int lane = threadIdx.x & 31;
    const _Float16* p = base + (size_t)(col + (lane & 15)) * ld + kb + ((lane >> 4) << 3);
    v16h r;
    ((h8*)&r)[0] = *(const h8*)(p);
    ((h8*)&r)[1] = *(const h8*)(p + 16);
    return r;
}
#define WMMA_F16(A, B, C) \
    __builtin_amdgcn_wmma_f32_16x16x32_f16(false, (A), false, (B), (short)0, (C), false, false)

// ---------------------------------------------------------------------------
// Kernel 1: fused Q/K/V projection + RoPE, f32 in -> f16 out.
// grid = (24, 128): x = p*8 + ntile (p in {q,k,v}), y = mtile (256 tokens).
// block = 256 threads = 8 waves; wave (wr,wc) computes 64x64 strip (one head).
// ---------------------------------------------------------------------------
__global__ __launch_bounds__(256) void qkv_gemm_kernel(
    const float* __restrict__ x,
    const float* __restrict__ wq, const float* __restrict__ wk, const float* __restrict__ wv,
    const float* __restrict__ bq, const float* __restrict__ bk, const float* __restrict__ bv,
    _Float16* __restrict__ qh, _Float16* __restrict__ kh, _Float16* __restrict__ vh)
{
    __shared__ __align__(16) _Float16 As[256 * 72];   // 256 rows x 64 K (+8 pad) = 36 KB
    __shared__ __align__(16) _Float16 Bs[128 * 72];   // 128 cols x 64 K (+8 pad) = 18 KB

    const int tid  = threadIdx.x;
    const int lane = tid & 31, wave = tid >> 5;
    const int p    = blockIdx.x >> 3;
    const int nb   = (blockIdx.x & 7) * 128;
    const int mb   = blockIdx.y * 256;

    const float* w    = (p == 0) ? wq : (p == 1) ? wk : wv;
    const float* bias = (p == 0) ? bq : (p == 1) ? bk : bv;
    _Float16*    outp = (p == 0) ? qh : (p == 1) ? kh : vh;

    const int wr = wave >> 1;   // 0..3 : 64-row strip
    const int wc = wave & 1;    // 0..1 : 64-col strip (one full head)

    v8f acc[4][4];
    for (int r = 0; r < 4; ++r)
        for (int c = 0; c < 4; ++c)
            for (int e = 0; e < 8; ++e) acc[r][c][e] = 0.f;

    for (int kb = 0; kb < D_MODEL; kb += 64) {
        // stage A: 256x64 f32 -> f16 (row-major, stride 72)
        for (int c = tid; c < 4096; c += 256) {
            int row = c >> 4, kc = (c & 15) << 2;
            float4 v4 = *(const float4*)(x + (size_t)(mb + row) * D_MODEL + kb + kc);
            h4 hv; hv[0] = (_Float16)v4.x; hv[1] = (_Float16)v4.y;
                   hv[2] = (_Float16)v4.z; hv[3] = (_Float16)v4.w;
            *(h4*)(As + row * 72 + kc) = hv;
        }
        // stage B transposed: Bs[n][k] = w[kb+k][nb+n], f32 -> f16
        for (int c = tid; c < 2048; c += 256) {
            int kk = c >> 5, nc = (c & 31) << 2;
            float4 v4 = *(const float4*)(w + (size_t)(kb + kk) * D_MODEL + nb + nc);
            Bs[(nc + 0) * 72 + kk] = (_Float16)v4.x;
            Bs[(nc + 1) * 72 + kk] = (_Float16)v4.y;
            Bs[(nc + 2) * 72 + kk] = (_Float16)v4.z;
            Bs[(nc + 3) * 72 + kk] = (_Float16)v4.w;
        }
        __syncthreads();

        for (int ks = 0; ks < 64; ks += 32) {
            v16h a[4];
            for (int rt = 0; rt < 4; ++rt)
                a[rt] = frag_a(As, wr * 64 + rt * 16, 72, ks);
            for (int ct = 0; ct < 4; ++ct) {
                v16h bf = frag_b(Bs, wc * 64 + ct * 16, 72, ks);
                for (int rt = 0; rt < 4; ++rt)
                    acc[rt][ct] = WMMA_F16(a[rt], bf, acc[rt][ct]);
            }
        }
        __syncthreads();
        if (kb + 64 < D_MODEL)
            __builtin_prefetch(x + (size_t)(mb + tid) * D_MODEL + kb + 64, 0, 1);
    }

    // epilogue: bias + RoPE (q,k only; position = window index), store f16
    const int lnn = lane & 15, hs = lane >> 4;
    // per-lane loop invariants: 4 bias values, 2 RoPE inverse frequencies
    float bv4[4];
    for (int ct = 0; ct < 4; ++ct) bv4[ct] = bias[nb + wc * 64 + ct * 16 + lnn];
    const float invf0 = __expf(-ROPE_LOGK * (float)lnn);          // f = lnn      (ct 0,2)
    const float invf1 = __expf(-ROPE_LOGK * (float)(16 + lnn));   // f = 16+lnn   (ct 1,3)

    for (int rt = 0; rt < 4; ++rt)
        for (int s = 0; s < 8; ++s) {
            int m  = mb + wr * 64 + rt * 16 + s + (hs << 3);     // global token
            int iw = (m & (SEQ_T - 1)) >> 6;                     // window index 0..127
            float sn0, cs0, sn1, cs1;
            __sincosf((float)iw * invf0, &sn0, &cs0);
            __sincosf((float)iw * invf1, &sn1, &cs1);
            for (int ct = 0; ct < 4; ++ct) {
                int col = nb + wc * 64 + ct * 16 + lnn;
                float v = acc[rt][ct][s] + bv4[ct];
                if (p < 2) {
                    float sn = (ct & 1) ? sn1 : sn0;
                    float cs = (ct & 1) ? cs1 : cs0;
                    float vp = acc[rt][ct ^ 2][s] + bv4[ct ^ 2];  // partner d^32
                    v = (ct < 2) ? (v * cs - vp * sn) : (v * cs + vp * sn);
                }
                outp[(size_t)m * D_MODEL + col] = (_Float16)v;
            }
        }
}

// ---------------------------------------------------------------------------
// Kernel 2: attention, one block per (window-batch bw, head h).
// Q,K,V: 128x64 f16. S = Q K^T /8 (+mask), softmax, O = P V. 4 waves.
// LDS: Vt[64][136] | Qs[128][72] | Ks[128][72]; P[128][136] overlaps Qs+Ks.
// ---------------------------------------------------------------------------
__global__ __launch_bounds__(128) void attn_kernel(
    const _Float16* __restrict__ qh, const _Float16* __restrict__ kh,
    const _Float16* __restrict__ vh, const unsigned char* __restrict__ pmask,
    _Float16* __restrict__ oh)
{
    __shared__ __align__(16) _Float16 smem[27136];   // 54272 bytes
    __shared__ float maskAdd[128];
    _Float16* Vt = smem;                 // [64][136]  V transposed (B layout for PV)
    _Float16* Qs = smem + 64 * 136;      // [128][72]
    _Float16* Ks = Qs + 128 * 72;        // [128][72]
    _Float16* Ps = Qs;                   // [128][136] reuses Q+K space after scores

    const int tid = threadIdx.x, lane = tid & 31, wave = tid >> 5;
    const int bw = blockIdx.x >> 4, h = blockIdx.x & 15;
    const int b = bw >> 6, j = bw & 63;

    // gather the dilated window: row i -> token b*8192 + i*64 + j
    for (int c = tid; c < 1024; c += 128) {
        int i = c >> 3, dc = (c & 7) << 3;
        size_t g = ((size_t)(b * SEQ_T + i * WDIL + j)) * D_MODEL + h * HEAD_DIM + dc;
        *(h8*)(Qs + i * 72 + dc) = *(const h8*)(qh + g);
        *(h8*)(Ks + i * 72 + dc) = *(const h8*)(kh + g);
        h8 vv = *(const h8*)(vh + g);
        for (int u = 0; u < 8; ++u) Vt[(dc + u) * 136 + i] = vv[u];
    }
    maskAdd[tid] = pmask[(size_t)b * SEQ_T + tid * WDIL + j] ? 0.f : -1e9f;
    __syncthreads();

    // S = Q K^T : wave owns rows [wave*32, wave*32+32), all 128 key columns
    v8f sacc[2][8];
    for (int r = 0; r < 2; ++r)
        for (int c = 0; c < 8; ++c)
            for (int e = 0; e < 8; ++e) sacc[r][c][e] = 0.f;
    for (int kb = 0; kb < HEAD_DIM; kb += 32) {
        v16h a0 = frag_a(Qs, wave * 32,      72, kb);
        v16h a1 = frag_a(Qs, wave * 32 + 16, 72, kb);
        for (int ct = 0; ct < 8; ++ct) {
            v16h bf = frag_b(Ks, ct * 16, 72, kb);
            sacc[0][ct] = WMMA_F16(a0, bf, sacc[0][ct]);
            sacc[1][ct] = WMMA_F16(a1, bf, sacc[1][ct]);
        }
    }
    __syncthreads();   // all waves done reading Q/K before P overwrites them

    // softmax in registers: C-tile row (m = s + 8*hs) lives across one 16-lane half
    const int lnn = lane & 15, hs = lane >> 4;
    for (int rt = 0; rt < 2; ++rt)
        for (int s = 0; s < 8; ++s) {
            float xv[8]; float mx = -3.4e38f;
            for (int ct = 0; ct < 8; ++ct) {
                xv[ct] = sacc[rt][ct][s] * 0.125f + maskAdd[ct * 16 + lnn];
                mx = fmaxf(mx, xv[ct]);
            }
            for (int off = 1; off < 16; off <<= 1) mx = fmaxf(mx, __shfl_xor(mx, off, 16));
            float sum = 0.f;
            for (int ct = 0; ct < 8; ++ct) { xv[ct] = __expf(xv[ct] - mx); sum += xv[ct]; }
            for (int off = 1; off < 16; off <<= 1) sum += __shfl_xor(sum, off, 16);
            float inv = 1.f / sum;
            int m = wave * 32 + rt * 16 + s + (hs << 3);
            for (int ct = 0; ct < 8; ++ct)
                Ps[m * 136 + ct * 16 + lnn] = (_Float16)(xv[ct] * inv);
        }
    __syncthreads();

    // O = P V
    v8f oacc[2][4];
    for (int r = 0; r < 2; ++r)
        for (int c = 0; c < 4; ++c)
            for (int e = 0; e < 8; ++e) oacc[r][c][e] = 0.f;
    for (int kb = 0; kb < NWIN; kb += 32) {
        v16h a0 = frag_a(Ps, wave * 32,      136, kb);
        v16h a1 = frag_a(Ps, wave * 32 + 16, 136, kb);
        for (int ct = 0; ct < 4; ++ct) {
            v16h bf = frag_b(Vt, ct * 16, 136, kb);
            oacc[0][ct] = WMMA_F16(a0, bf, oacc[0][ct]);
            oacc[1][ct] = WMMA_F16(a1, bf, oacc[1][ct]);
        }
    }

    // scatter back to token layout (undoes windowing for the O-projection)
    for (int rt = 0; rt < 2; ++rt)
        for (int s = 0; s < 8; ++s) {
            int m = wave * 32 + rt * 16 + s + (hs << 3);   // query window index
            size_t g = ((size_t)(b * SEQ_T + m * WDIL + j)) * D_MODEL + h * HEAD_DIM;
            for (int ct = 0; ct < 4; ++ct)
                oh[g + ct * 16 + lnn] = (_Float16)oacc[rt][ct][s];
        }
}

// ---------------------------------------------------------------------------
// Kernel 3: output projection, f16 A @ (f32->f16) wo + bo -> f32 out (B,T,C).
// grid = (8, 128), block = 256; 256x128 block tile, 64x64 wave tile.
// ---------------------------------------------------------------------------
__global__ __launch_bounds__(256) void out_gemm_kernel(
    const _Float16* __restrict__ ah, const float* __restrict__ wo,
    const float* __restrict__ bo, float* __restrict__ out)
{
    __shared__ __align__(16) _Float16 As[256 * 72];
    __shared__ __align__(16) _Float16 Bs[128 * 72];

    const int tid  = threadIdx.x;
    const int lane = tid & 31, wave = tid >> 5;
    const int nb   = blockIdx.x * 128;
    const int mb   = blockIdx.y * 256;
    const int wr = wave >> 1, wc = wave & 1;

    v8f acc[4][4];
    for (int r = 0; r < 4; ++r)
        for (int c = 0; c < 4; ++c)
            for (int e = 0; e < 8; ++e) acc[r][c][e] = 0.f;

    for (int kb = 0; kb < D_MODEL; kb += 64) {
        for (int c = tid; c < 2048; c += 256) {           // A already f16: 256x64
            int row = c >> 3, kc = (c & 7) << 3;
            *(h8*)(As + row * 72 + kc) =
                *(const h8*)(ah + (size_t)(mb + row) * D_MODEL + kb + kc);
        }
        for (int c = tid; c < 2048; c += 256) {           // wo transposed f32->f16: 64x128
            int kk = c >> 5, nc = (c & 31) << 2;
            float4 v4 = *(const float4*)(wo + (size_t)(kb + kk) * D_MODEL + nb + nc);
            Bs[(nc + 0) * 72 + kk] = (_Float16)v4.x;
            Bs[(nc + 1) * 72 + kk] = (_Float16)v4.y;
            Bs[(nc + 2) * 72 + kk] = (_Float16)v4.z;
            Bs[(nc + 3) * 72 + kk] = (_Float16)v4.w;
        }
        __syncthreads();

        for (int ks = 0; ks < 64; ks += 32) {
            v16h a[4];
            for (int rt = 0; rt < 4; ++rt)
                a[rt] = frag_a(As, wr * 64 + rt * 16, 72, ks);
            for (int ct = 0; ct < 4; ++ct) {
                v16h bf = frag_b(Bs, wc * 64 + ct * 16, 72, ks);
                for (int rt = 0; rt < 4; ++rt)
                    acc[rt][ct] = WMMA_F16(a[rt], bf, acc[rt][ct]);
            }
        }
        __syncthreads();
        if (kb + 64 < D_MODEL)
            __builtin_prefetch(ah + (size_t)(mb + tid) * D_MODEL + kb + 64, 0, 1);
    }

    const int lnn = lane & 15, hs = lane >> 4;
    float bv4[4];
    for (int ct = 0; ct < 4; ++ct) bv4[ct] = bo[nb + wc * 64 + ct * 16 + lnn];
    for (int rt = 0; rt < 4; ++rt)
        for (int s = 0; s < 8; ++s) {
            int m = mb + wr * 64 + rt * 16 + s + (hs << 3);
            for (int ct = 0; ct < 4; ++ct) {
                int col = nb + wc * 64 + ct * 16 + lnn;
                out[(size_t)m * D_MODEL + col] = acc[rt][ct][s] + bv4[ct];
            }
        }
}

// ---------------------------------------------------------------------------
extern "C" void kernel_launch(void* const* d_in, const int* in_sizes, int n_in,
                              void* d_out, int out_size, void* d_ws, size_t ws_size,
                              hipStream_t stream) {
    const float*         x     = (const float*)d_in[0];
    const unsigned char* pmask = (const unsigned char*)d_in[1];
    // d_in[2]=cos, d_in[3]=sin : unused — reference discards them (RoPE uses window index)
    const float* wq = (const float*)d_in[4];
    const float* bq = (const float*)d_in[5];
    const float* wk = (const float*)d_in[6];
    const float* bk = (const float*)d_in[7];
    const float* wv = (const float*)d_in[8];
    const float* bv = (const float*)d_in[9];
    const float* wo = (const float*)d_in[10];
    const float* bo = (const float*)d_in[11];
    float* out = (float*)d_out;

    const size_t NE = (size_t)NTOK * D_MODEL;    // 33.5M elements (64 MiB as f16)
    _Float16* qh  = (_Float16*)d_ws;
    _Float16* kh  = qh + NE;
    _Float16* vh  = kh + NE;
    _Float16* ahb = vh + NE;                     // attention output, pre-O-projection

    qkv_gemm_kernel<<<dim3(24, 128), 256, 0, stream>>>(x, wq, wk, wv, bq, bk, bv, qh, kh, vh);
    attn_kernel<<<dim3(4096), 128, 0, stream>>>(qh, kh, vh, pmask, ahb);
    out_gemm_kernel<<<dim3(8, 128), 256, 0, stream>>>(ahb, wo, bo, out);
}
